// RandomWalkProbabilityGenerator_68436008894838
// MI455X (gfx1250) — matrix operand order repaired and checked by reading, not compiled, not run on hardware
//
#include <hip/hip_runtime.h>
#include <hip/hip_bf16.h>
#include <stdint.h>
#include <stddef.h>

// ---------------- CDNA5 WMMA types ----------------
typedef __attribute__((ext_vector_type(16))) __bf16 bf16x16;
typedef __attribute__((ext_vector_type(8)))  float  f32x8;

#define NNODES 8192   // fixed by the harness setup (N=8192)

__device__ __forceinline__ float bf2f(unsigned short u) {
  union { unsigned int i; float f; } c; c.i = ((unsigned int)u) << 16; return c.f;
}
__device__ __forceinline__ unsigned short f2bf(float f) {
  union { float f; unsigned int i; } c; c.f = f;
  unsigned int r = c.i + 0x7FFFu + ((c.i >> 16) & 1u);  // round-to-nearest-even
  return (unsigned short)(r >> 16);
}

// ---------------- 1) weighted out-degree ----------------
__global__ void deg_kernel(const int* __restrict__ src,
                           const float* __restrict__ w,
                           float* __restrict__ deg, int E) {
  int t = blockIdx.x * blockDim.x + threadIdx.x;
  if (t < E) atomicAdd(&deg[src[t]], w[t]);
}

// ---------------- 2) scatter P and P^T (bf16) ----------------
__global__ void scatter_kernel(const int* __restrict__ ei,   // [2*E]: src then dst
                               const float* __restrict__ w,
                               const float* __restrict__ deg,
                               unsigned short* __restrict__ P,
                               unsigned short* __restrict__ PT,
                               int E, int N) {
  int t = blockIdx.x * blockDim.x + threadIdx.x;
  if (t >= E) return;
  int s = ei[t];
  int d = ei[t + E];
  float v = w[t] / fmaxf(deg[s], 1e-8f);
  unsigned short b = f2bf(v);
  P [(size_t)s * N + d] = b;   // duplicate edges: some write wins (matches scatter-set)
  PT[(size_t)d * N + s] = b;
}

// ---------------- 3) diag(P) and diag(P^2) ----------------
__global__ void diag12_kernel(const unsigned short* __restrict__ P,
                              const unsigned short* __restrict__ PT,
                              float* __restrict__ d1, float* __restrict__ d2, int N) {
  int lane = threadIdx.x & 31;
  int wid  = threadIdx.x >> 5;
  int row  = blockIdx.x * 8 + wid;
  if (row >= N) return;
  const uint4* pr = (const uint4*)(P  + (size_t)row * N);
  const uint4* pt = (const uint4*)(PT + (size_t)row * N);
  float s = 0.f;
  for (int it = lane; it < N / 8; it += 32) {
    union { uint4 q; unsigned short h[8]; } ua, ub;
    ua.q = pr[it]; ub.q = pt[it];
#pragma unroll
    for (int e = 0; e < 8; ++e) s += bf2f(ua.h[e]) * bf2f(ub.h[e]);
  }
#pragma unroll
  for (int off = 16; off >= 1; off >>= 1) s += __shfl_xor(s, off, 32);
  if (lane == 0) {
    d2[row] = s;
    d1[row] = bf2f(P[(size_t)row * N + row]);
  }
}

// ---------------- 4) LDS-tiled fused GEMM (P^2 = P*P) + diag(P^3) ----------
// Workgroup: 256 threads / 8 waves, block tile 128x128, BK=32.
// LDS holds the A block (P rows) and B block (PT rows) at an 80B row pitch
// (32 data bf16 + 8 pad) so 16-lane fragment reads at 80B stride hit 16
// distinct bank groups. Double-buffered; one barrier per K-step.
// Each wave computes 32x64 = 8 v_wmma_f32_16x16x32_bf16 tiles per K-step.
// diag3 fused in epilogue; P^2 never touches memory.
#define BM 128
#define BN 128
#define BK 32
#define LPITCH 40   // bf16 elements per LDS row (80 bytes)

__global__ void __launch_bounds__(256)
gemm_diag3_kernel(const unsigned short* __restrict__ P,
                  const unsigned short* __restrict__ PT,
                  float* __restrict__ d3, int N) {
  __shared__ unsigned short As[2][BM * LPITCH];
  __shared__ unsigned short Bs[2][BM * LPITCH];

  const int tid  = threadIdx.x;
  const int lane = tid & 31;
  const int wid  = tid >> 5;
  const int l16  = lane & 15;
  const int half = lane >> 4;

  const int nbx = N / BN;                       // 64
  const int bm0 = (blockIdx.x / nbx) * BM;
  const int bn0 = (blockIdx.x % nbx) * BN;

  const int wm = wid & 3;                       // 0..3 : 32-row group
  const int wn = wid >> 2;                      // 0..1 : 64-col group

  // ---- staging map: 512 uint4 chunks per matrix, 2 per thread ----
  // chunk q: row = q>>2 (0..127), 16B piece = q&3
  const int r0 = tid >> 2;                      // rows 0..63   (q = tid)
  const int r1 = r0 + 64;                       // rows 64..127 (q = tid+256)
  const int c8 = (tid & 3) * 8;                 // element offset of 16B piece

  const unsigned short* gA0 = P  + (size_t)(bm0 + r0) * N + c8;
  const unsigned short* gA1 = P  + (size_t)(bm0 + r1) * N + c8;
  const unsigned short* gB0 = PT + (size_t)(bn0 + r0) * N + c8;
  const unsigned short* gB1 = PT + (size_t)(bn0 + r1) * N + c8;

  const int sOffA0 = r0 * LPITCH + c8;
  const int sOffA1 = r1 * LPITCH + c8;

  f32x8 acc[2][4];
#pragma unroll
  for (int s = 0; s < 2; ++s)
#pragma unroll
    for (int t = 0; t < 4; ++t)
#pragma unroll
      for (int r = 0; r < 8; ++r) acc[s][t][r] = 0.f;

  // ---- prologue: stage K-block 0 into buffer 0 ----
  {
    uint4 a0 = *(const uint4*)(gA0);
    uint4 a1 = *(const uint4*)(gA1);
    uint4 b0 = *(const uint4*)(gB0);
    uint4 b1 = *(const uint4*)(gB1);
    *(uint4*)(&As[0][sOffA0]) = a0;
    *(uint4*)(&As[0][sOffA1]) = a1;
    *(uint4*)(&Bs[0][sOffA0]) = b0;
    *(uint4*)(&Bs[0][sOffA1]) = b1;
  }
  __syncthreads();

  for (int kb = 0; kb < N; kb += BK) {
    const int cur = (kb >> 5) & 1;
    const int nxt = cur ^ 1;
    const bool more = (kb + BK) < N;

    // issue next K-block's global loads early (overlap with LDS-fed WMMAs)
    uint4 a0 = {}, a1 = {}, b0 = {}, b1 = {};
    if (more) {
      a0 = *(const uint4*)(gA0 + kb + BK);
      a1 = *(const uint4*)(gA1 + kb + BK);
      b0 = *(const uint4*)(gB0 + kb + BK);
      b1 = *(const uint4*)(gB1 + kb + BK);
      if (kb + 2 * BK < N) {                    // prefetch one stage further
        __builtin_prefetch(gA0 + kb + 2 * BK, 0, 3);
        __builtin_prefetch(gB0 + kb + 2 * BK, 0, 3);
      }
    }

    // ---- compute from LDS[cur]: 8 WMMA tiles ----
    bf16x16 afrag[2];
#pragma unroll
    for (int s = 0; s < 2; ++s) {
      // A fragment: lane row = wm*32 + s*16 + l16;
      // elems 0..7 at K=half*8, elems 8..15 at K=half*8+16
      const int row = wm * 32 + s * 16 + l16;
      union { uint4 q[2]; bf16x16 v; } u;
      u.q[0] = *(const uint4*)(&As[cur][row * LPITCH + half * 8]);
      u.q[1] = *(const uint4*)(&As[cur][row * LPITCH + half * 8 + 16]);
      afrag[s] = u.v;
    }
#pragma unroll
    for (int t = 0; t < 4; ++t) {
      // B fragment: lane col = wn*64 + t*16 + l16; elems 0..7 at K=half*8, elems 8..15 at K=half*8+16
      const int row = wn * 64 + t * 16 + l16;
      union { uint4 q[2]; bf16x16 v; } u;
      u.q[0] = *(const uint4*)(&Bs[cur][row * LPITCH + half * 8]);
      u.q[1] = *(const uint4*)(&Bs[cur][row * LPITCH + half * 8 + 16]);
#pragma unroll
      for (int s = 0; s < 2; ++s) {
        acc[s][t] = __builtin_amdgcn_wmma_f32_16x16x32_bf16(
            false, afrag[s], false, u.v, (short)0, acc[s][t], false, false);
      }
    }

    // ---- store next K-block into LDS[nxt] ----
    if (more) {
      *(uint4*)(&As[nxt][sOffA0]) = a0;
      *(uint4*)(&As[nxt][sOffA1]) = a1;
      *(uint4*)(&Bs[nxt][sOffA0]) = b0;
      *(uint4*)(&Bs[nxt][sOffA1]) = b1;
    }
    __syncthreads();   // single barrier: cur-reads done before nxt becomes cur
  }

  // ---- epilogue: diag3[i] += sum_j P2[i,j]*PT[i,j] over this 32x64 patch --
  // C/D layout: VGPR r, lanes 0-15 -> local row +r; lanes 16-31 -> +8+r;
  //             col = t*16 + (lane&15).
#pragma unroll
  for (int s = 0; s < 2; ++s) {
    float rowsum[8];
#pragma unroll
    for (int r = 0; r < 8; ++r) rowsum[r] = 0.f;
#pragma unroll
    for (int t = 0; t < 4; ++t) {
#pragma unroll
      for (int r = 0; r < 8; ++r) {
        int i = bm0 + wm * 32 + s * 16 + r + half * 8;
        int j = bn0 + wn * 64 + t * 16 + l16;
        rowsum[r] += acc[s][t][r] * bf2f(PT[(size_t)i * N + j]);
      }
    }
#pragma unroll
    for (int r = 0; r < 8; ++r) {
      float v = rowsum[r];
      v += __shfl_xor(v, 1, 32);
      v += __shfl_xor(v, 2, 32);
      v += __shfl_xor(v, 4, 32);
      v += __shfl_xor(v, 8, 32);
      if (l16 == 0)
        atomicAdd(&d3[bm0 + wm * 32 + s * 16 + r + half * 8], v);
    }
  }
}

// ---------------- 5) finalize mu / variance ----------------
__global__ void finalize_kernel(const float* __restrict__ d1,
                                const float* __restrict__ d2,
                                const float* __restrict__ d3,
                                float* __restrict__ out, int N) {
  int i = blockIdx.x * blockDim.x + threadIdx.x;
  if (i >= N) return;
  float a = d1[i], b = d2[i], c = d3[i];
  float mu = a + b + c;
  out[i]     = mu;
  out[N + i] = mu - (a * a + b * b + c * c);
}

extern "C" void kernel_launch(void* const* d_in, const int* in_sizes, int n_in,
                              void* d_out, int out_size, void* d_ws, size_t ws_size,
                              hipStream_t stream) {
  const int N = NNODES;
  const int E = in_sizes[0] / 2;
  const int*   ei = (const int*)d_in[0];
  const float* ew = (const float*)d_in[1];

  // Workspace layout
  unsigned short* Pbf  = (unsigned short*)d_ws;                 // N*N bf16
  unsigned short* PTbf = Pbf + (size_t)N * N;                   // N*N bf16
  float* deg = (float*)(PTbf + (size_t)N * N);                  // N f32
  float* d1  = deg + N;
  float* d2  = d1 + N;
  float* d3  = d2 + N;

  hipMemsetAsync(Pbf,  0, (size_t)N * N * sizeof(unsigned short), stream);
  hipMemsetAsync(PTbf, 0, (size_t)N * N * sizeof(unsigned short), stream);
  hipMemsetAsync(deg,  0, (size_t)N * sizeof(float), stream);
  hipMemsetAsync(d3,   0, (size_t)N * sizeof(float), stream);

  deg_kernel<<<(E + 255) / 256, 256, 0, stream>>>(ei, ew, deg, E);
  scatter_kernel<<<(E + 255) / 256, 256, 0, stream>>>(ei, ew, deg, Pbf, PTbf, E, N);
  diag12_kernel<<<N / 8, 256, 0, stream>>>(Pbf, PTbf, d1, d2, N);
  // (8192/128)^2 = 4096 blocks of 8 waves, each block = 128x128 of P^2
  gemm_diag3_kernel<<<(N / BM) * (N / BN), 256, 0, stream>>>(Pbf, PTbf, d3, N);
  finalize_kernel<<<N / 256, 256, 0, stream>>>(d1, d2, d3, (float*)d_out, N);
}
